// EdgePredictionModel_46583215292497
// MI455X (gfx1250) — matrix operand order repaired and checked by reading, not compiled, not run on hardware
//
#include <hip/hip_runtime.h>
#include <hip/hip_bf16.h>

typedef __attribute__((ext_vector_type(16))) _Float16 v16h;
typedef __attribute__((ext_vector_type(8)))  float    v8f;

// ---------------------------------------------------------------------------
// A-fragment loader for V_WMMA_F32_16X16X32_F16 (16x32 f16 A, row = lane%16).
// ISA layout: lanes 0-15 hold K = [0..7] and [16..23]; lanes 16-31 hold
// K = [8..15] and [24..31]  (kb = 8*(lane>=16)); kf selects K-block 0/1 (+32).
// ---------------------------------------------------------------------------
template<bool RELU>
__device__ __forceinline__ v16h load_a_frag(const float* __restrict__ rowp,
                                            int kf, int hh) {
  const int kb = 8 * hh + 32 * kf;
  const float4* q = (const float4*)(rowp + kb);
  float4 us[4];
  us[0] = q[0]; us[1] = q[1];   // K = kb .. kb+7
  us[2] = q[4]; us[3] = q[5];   // K = kb+16 .. kb+23
  v16h a;
#pragma unroll
  for (int p = 0; p < 4; ++p) {
    float v0 = us[p].x, v1 = us[p].y, v2 = us[p].z, v3 = us[p].w;
    if (RELU) {
      v0 = fmaxf(v0, 0.f); v1 = fmaxf(v1, 0.f);
      v2 = fmaxf(v2, 0.f); v3 = fmaxf(v3, 0.f);
    }
    a[4*p + 0] = (_Float16)v0; a[4*p + 1] = (_Float16)v1;
    a[4*p + 2] = (_Float16)v2; a[4*p + 3] = (_Float16)v3;
  }
  return a;
}

// ---------------------------------------------------------------------------
// Zero-init workspace accumulators.
// ---------------------------------------------------------------------------
__global__ void zero_kernel(float* __restrict__ p, long long n) {
  long long i = (long long)blockIdx.x * blockDim.x + threadIdx.x;
  long long s = (long long)gridDim.x * blockDim.x;
  for (; i < n; i += s) p[i] = 0.f;
}

// ---------------------------------------------------------------------------
// C[nrows,64] = act(A[nrows,64]) @ W[64,64], one wave per 16-row tile.
// W pre-swizzled to LDS in B-fragment order: [ct][kf][lane][16 halves].
// ---------------------------------------------------------------------------
template<bool RELU_IN>
__global__ __launch_bounds__(256)
void gemm_n64_kernel(const float* __restrict__ A, const float* __restrict__ W,
                     float* __restrict__ C, int nrows) {
  __shared__ __align__(32) _Float16 wB[4 * 2 * 32 * 16];   // 8 KB
  for (int i = threadIdx.x; i < 4096; i += blockDim.x) {
    int k  = i & 15;
    int ln = (i >> 4) & 31;
    int kf = (i >> 9) & 1;
    int ct = i >> 10;
    int K  = (k & 7) + 8 * (ln >> 4) + 16 * (k >> 3) + 32 * kf;   // B layout: lane half selects K+8, elems 8-15 at K+16
    int Nc = ct * 16 + (ln & 15);
    wB[i] = (_Float16)W[K * 64 + Nc];
  }
  __syncthreads();

  const int wave = threadIdx.x >> 5;
  const int lane = threadIdx.x & 31;
  const int hh   = lane >> 4;
  const int ntiles = (nrows + 15) >> 4;
  const int tile = blockIdx.x * (blockDim.x >> 5) + wave;
  if (tile >= ntiles) return;

  int arow = tile * 16 + (lane & 15);
  if (arow > nrows - 1) arow = nrows - 1;         // clamp tail gathers
  const float* rowp = A + (long long)arow * 64;
  v16h a0 = load_a_frag<RELU_IN>(rowp, 0, hh);
  v16h a1 = load_a_frag<RELU_IN>(rowp, 1, hh);

#pragma unroll
  for (int ct = 0; ct < 4; ++ct) {
    v16h b0 = *(const v16h*)&wB[((ct * 2 + 0) * 32 + lane) * 16];
    v16h b1 = *(const v16h*)&wB[((ct * 2 + 1) * 32 + lane) * 16];
    v8f c = {};
    c = __builtin_amdgcn_wmma_f32_16x16x32_f16(false, a0, false, b0,
                                               (short)0, c, false, false);
    c = __builtin_amdgcn_wmma_f32_16x16x32_f16(false, a1, false, b1,
                                               (short)0, c, false, false);
#pragma unroll
    for (int r = 0; r < 8; ++r) {
      int grow = tile * 16 + r + 8 * hh;          // C layout: M = r + 8*half
      if (grow < nrows)
        C[(long long)grow * 64 + ct * 16 + (lane & 15)] = c[r];
    }
  }
}

// ---------------------------------------------------------------------------
// acc[to[e]] += norm[e] * h[from[e]]  over all (edge, col) pairs.
// ---------------------------------------------------------------------------
__global__ __launch_bounds__(256)
void scatter_kernel(const float* __restrict__ h, const float* __restrict__ norm,
                    const int* __restrict__ from_idx, const int* __restrict__ to_idx,
                    float* __restrict__ acc, long long total) {
  long long i = (long long)blockIdx.x * blockDim.x + threadIdx.x;
  long long s = (long long)gridDim.x * blockDim.x;
  for (; i < total; i += s) {
    int e = (int)(i >> 6);
    int c = (int)(i & 63);
    float v = norm[e] * h[(long long)from_idx[e] * 64 + c];
    atomicAdd(&acc[(long long)to_idx[e] * 64 + c], v);   // global_atomic_add_f32
  }
}

// ---------------------------------------------------------------------------
// Edge predictor: out[e] = relu(relu(x2[src[e]]) @ W1 + b1) @ w2 + b2
// One wave per 16-edge tile: 16 col-tiles x 2 WMMAs, layer-2 dot fused into
// the epilogue, cross-lane reduced with shfl_xor.
// ---------------------------------------------------------------------------
__global__ __launch_bounds__(256)
void edge_pred_kernel(const float* __restrict__ x2, const int* __restrict__ src,
                      const float* __restrict__ W1, const float* __restrict__ b1,
                      const float* __restrict__ w2, const float* __restrict__ b2,
                      float* __restrict__ out, int E_) {
  __shared__ __align__(32) _Float16 w1B[16 * 2 * 32 * 16];  // 32 KB, frag order
  __shared__ float sb1[256];
  __shared__ float sw2[256];

  for (int i = threadIdx.x; i < 16384; i += blockDim.x) {
    int k  = i & 15;
    int ln = (i >> 4) & 31;
    int kf = (i >> 9) & 1;
    int ct = i >> 10;
    int K  = (k & 7) + 8 * (ln >> 4) + 16 * (k >> 3) + 32 * kf;
    int Nc = ct * 16 + (ln & 15);
    w1B[i] = (_Float16)W1[K * 256 + Nc];
  }
  for (int i = threadIdx.x; i < 256; i += blockDim.x) {
    sb1[i] = b1[i];
    sw2[i] = w2[i];
  }
  __syncthreads();

  const float bias2 = b2[0];
  const int wave = threadIdx.x >> 5;
  const int lane = threadIdx.x & 31;
  const int hh   = lane >> 4;
  const int ntiles = (E_ + 15) >> 4;
  const int nwaves = gridDim.x * (blockDim.x >> 5);

  for (int tile = blockIdx.x * (blockDim.x >> 5) + wave; tile < ntiles;
       tile += nwaves) {
    const int e0 = tile * 16;
    int eidx = e0 + (lane & 15);
    if (eidx > E_ - 1) eidx = E_ - 1;             // clamp tail gathers
    const float* rowp = x2 + (long long)src[eidx] * 64;
    v16h a0 = load_a_frag<true>(rowp, 0, hh);     // relu fused into load
    v16h a1 = load_a_frag<true>(rowp, 1, hh);

    float p[8];
#pragma unroll
    for (int r = 0; r < 8; ++r) p[r] = 0.f;

#pragma unroll
    for (int ct = 0; ct < 16; ++ct) {
      v16h b0 = *(const v16h*)&w1B[((ct * 2 + 0) * 32 + lane) * 16];
      v16h b1f = *(const v16h*)&w1B[((ct * 2 + 1) * 32 + lane) * 16];
      v8f c = {};
      c = __builtin_amdgcn_wmma_f32_16x16x32_f16(false, a0, false, b0,
                                                 (short)0, c, false, false);
      c = __builtin_amdgcn_wmma_f32_16x16x32_f16(false, a1, false, b1f,
                                                 (short)0, c, false, false);
      const int col = ct * 16 + (lane & 15);
      const float bb = sb1[col];
      const float ww = sw2[col];
#pragma unroll
      for (int r = 0; r < 8; ++r)
        p[r] += fmaxf(c[r] + bb, 0.f) * ww;       // layer 2 fused
    }

    // Reduce over the 16 lanes of each half (each half owns 8 edge rows).
#pragma unroll
    for (int m = 1; m < 16; m <<= 1) {
#pragma unroll
      for (int r = 0; r < 8; ++r) p[r] += __shfl_xor(p[r], m, 32);
    }
    if ((lane & 15) == 0) {
#pragma unroll
      for (int r = 0; r < 8; ++r) {
        int e = e0 + r + 8 * hh;
        if (e < E_) out[e] = p[r] + bias2;
      }
    }
  }
}

// ---------------------------------------------------------------------------
// Launch: zero accs -> GEMM(v2e) -> scatter(src->dst) -> GEMM(e2v, relu-in)
//         -> scatter(dst->src) -> fused edge predictor.
// ---------------------------------------------------------------------------
extern "C" void kernel_launch(void* const* d_in, const int* in_sizes, int n_in,
                              void* d_out, int out_size, void* d_ws, size_t ws_size,
                              hipStream_t stream) {
  const float* x    = (const float*)d_in[0];
  const float* norm = (const float*)d_in[1];
  const float* Wv2e = (const float*)d_in[2];
  const float* We2v = (const float*)d_in[3];
  const float* w1   = (const float*)d_in[4];
  const float* b1   = (const float*)d_in[5];
  const float* w2   = (const float*)d_in[6];
  const float* b2   = (const float*)d_in[7];
  const int*   src  = (const int*)d_in[8];
  const int*   dst  = (const int*)d_in[9];

  const int E_ = in_sizes[8];          // 2,000,000
  const int ND = in_sizes[0];          // N * 64
  const int Nn = ND / 64;              // 150,000

  float* acc1 = (float*)d_ws;          // [N,64]
  float* acc2 = acc1 + ND;             // [N,64]
  float* hbuf = acc2 + ND;             // [N,64] reused for both GEMM outputs

  const long long scatter_total = (long long)E_ * 64;
  const int gemm_blocks = (((Nn + 15) / 16) + 7) / 8;   // 8 waves/block

  zero_kernel<<<8192, 256, 0, stream>>>(acc1, (long long)2 * ND);

  gemm_n64_kernel<false><<<gemm_blocks, 256, 0, stream>>>(x, Wv2e, hbuf, Nn);
  scatter_kernel<<<32768, 256, 0, stream>>>(hbuf, norm, src, dst, acc1,
                                            scatter_total);
  gemm_n64_kernel<true><<<gemm_blocks, 256, 0, stream>>>(acc1, We2v, hbuf, Nn);
  scatter_kernel<<<32768, 256, 0, stream>>>(hbuf, norm, dst, src, acc2,
                                            scatter_total);
  edge_pred_kernel<<<4096, 256, 0, stream>>>(acc2, src, w1, b1, w2, b2,
                                             (float*)d_out, E_);
}
